// AXSLinearUnified_19696720020171
// MI455X (gfx1250) — matrix operand order repaired and checked by reading, not compiled, not run on hardware
//
#include <hip/hip_runtime.h>

typedef _Float16 v16h __attribute__((ext_vector_type(16)));
typedef _Float16 v8h  __attribute__((ext_vector_type(8)));
typedef _Float16 v4h  __attribute__((ext_vector_type(4)));
typedef float    v8f  __attribute__((ext_vector_type(8)));
typedef float    v4f  __attribute__((ext_vector_type(4)));
typedef int      gv4i __attribute__((vector_size(16)));   // matches async-builtin param type

#define TM 128
#define TN 128
#define TK 32
#define LDS_STRIDE 40   // 32 + 8 pad f16 (=80B: keeps 16B alignment, kills bank conflicts)
#define KDIM 1024
#define NDIM 1024
#define NF5_OFFSET 0.9677083f

#if defined(__gfx1250__) && __has_builtin(__builtin_amdgcn_global_load_async_to_lds_b128)
#define USE_ASYNC_LDS 1
#else
#define USE_ASYNC_LDS 0
#endif

#define AS1 __attribute__((address_space(1)))
#define AS3 __attribute__((address_space(3)))

// ---------------------------------------------------------------------------
// NF5 block quantize f32 -> f16 (dequantized values), block size 32.
// 4 elements per lane: one wave = 128 elements = 4 NF5 blocks (8 lanes each).
// amax: 3 in-lane fmax + 3 shfl_xor within the 8-lane group.
// searchsorted: branchless 5-step binary search over 31 LDS boundaries.
// ---------------------------------------------------------------------------
__global__ __launch_bounds__(256)
void nf5_quant_f32_to_f16(const float4* __restrict__ in, v4h* __restrict__ out,
                          long long n4)      // n4 = n/4
{
    __shared__ float s_tbl[32];
    __shared__ float s_bnd[31];
    const int tid = threadIdx.x;

    if (tid < 32) {
        float t;
        if (tid < 16) {
            const float p0 = 1.0f - NF5_OFFSET;
            float p = p0 + (0.5f - p0) * ((float)tid * (1.0f / 16.0f));
            float denom = -(1.41421356237f * erfinvf(2.0f * p0 - 1.0f));
            t = (1.41421356237f * erfinvf(2.0f * p - 1.0f)) / denom;
        } else if (tid == 16) {
            t = 0.0f;
        } else {
            int j = tid - 16;
            float p = 0.5f + (NF5_OFFSET - 0.5f) * ((float)j * (1.0f / 15.0f));
            float denom = 1.41421356237f * erfinvf(2.0f * NF5_OFFSET - 1.0f);
            t = (1.41421356237f * erfinvf(2.0f * p - 1.0f)) / denom;
        }
        s_tbl[tid] = t;
    }
    __syncthreads();
    if (tid < 31) s_bnd[tid] = 0.5f * (s_tbl[tid] + s_tbl[tid + 1]);
    __syncthreads();

    // rank(nv) = #{ boundaries < nv }  (searchsorted side='left')
    auto rank31 = [&](float nv) -> int {
        int q = 0;
        q  = (s_bnd[15]    < nv) ? 16 : 0;
        q += (s_bnd[q + 7] < nv) ? 8 : 0;
        q += (s_bnd[q + 3] < nv) ? 4 : 0;
        q += (s_bnd[q + 1] < nv) ? 2 : 0;
        q += (s_bnd[q]     < nv) ? 1 : 0;
        return q;
    };

    long long i0     = (long long)blockIdx.x * blockDim.x + tid;
    long long stride = (long long)gridDim.x * blockDim.x;
    for (long long e = i0; e < n4; e += stride) {
        float4 xv = in[e];
        // block amax: 4 in-lane values, then reduce over the 8 lanes of this block
        float m = fmaxf(fmaxf(fabsf(xv.x), fabsf(xv.y)),
                        fmaxf(fabsf(xv.z), fabsf(xv.w)));
        m = fmaxf(m, __shfl_xor(m, 1, 32));
        m = fmaxf(m, __shfl_xor(m, 2, 32));
        m = fmaxf(m, __shfl_xor(m, 4, 32));
        m = fmaxf(m, 1e-12f);
        // power-of-2 scale: exp2(ceil(log2(m))); reciprocal is exact
        int ex;
        float mant   = frexpf(m, &ex);            // m = mant * 2^ex, mant in [0.5, 1)
        int   e2     = (mant == 0.5f) ? (ex - 1) : ex;
        float scale  = ldexpf(1.0f, e2);
        float rscale = ldexpf(1.0f, -e2);

        v4h o;
        o[0] = (_Float16)(s_tbl[rank31(xv.x * rscale)] * scale);
        o[1] = (_Float16)(s_tbl[rank31(xv.y * rscale)] * scale);
        o[2] = (_Float16)(s_tbl[rank31(xv.z * rscale)] * scale);
        o[3] = (_Float16)(s_tbl[rank31(xv.w * rscale)] * scale);
        out[e] = o;
    }
}

// ---------------------------------------------------------------------------
// f16 WMMA GEMM: C[M,N] = A[M,K] * B[N,K]^T + bias, f32 accumulate
// 256 threads = 8 waves (4 x 2), block tile 128x128, k-tile 32, double-buffered LDS
// Staging uses GLOBAL_LOAD_ASYNC_TO_LDS_B128 (ASYNCcnt) when available.
// ---------------------------------------------------------------------------
__global__ __launch_bounds__(256)
void nf5_wmma_gemm(const _Float16* __restrict__ A,   // [M, K] quantized x
                   const _Float16* __restrict__ B,   // [N, K] quantized weight
                   const float*    __restrict__ bias,
                   float*          __restrict__ C,   // [M, N]
                   int M)
{
    __shared__ _Float16 As[2][TM * LDS_STRIDE];
    __shared__ _Float16 Bs[2][TN * LDS_STRIDE];

    const int tid  = threadIdx.x;
    const int lane = tid & 31;
    const int wid  = tid >> 5;           // 0..7
    const int wm   = wid & 3;            // wave row tile (32 rows each)
    const int wn   = wid >> 2;           // wave col tile (64 cols each)
    const int r    = lane & 15;
    const int hs   = lane >> 4;          // half-wave select

    const int mBlock = blockIdx.y * TM;
    const int nBlock = blockIdx.x * TN;

    // staging: 512 x 16B chunks per tile -> 2 per thread per matrix
    const int rlo = tid >> 2;            // rows 0..63
    const int rhi = rlo + 64;            // rows 64..127
    const int kq  = (tid & 3) * 8;       // f16 offset within k-tile row: 0/8/16/24

    const int NK = KDIM / TK;            // 32 k-tiles

#if USE_ASYNC_LDS
    // --- direct global -> LDS async copy (no VGPR staging) ------------------
    auto stage = [&](int kt, int buf) {
        const int kb = kt * TK + kq;
        __builtin_amdgcn_global_load_async_to_lds_b128(
            (AS1 gv4i*)(A + (size_t)(mBlock + rlo) * KDIM + kb),
            (AS3 gv4i*)&As[buf][rlo * LDS_STRIDE + kq], 0, 0);
        __builtin_amdgcn_global_load_async_to_lds_b128(
            (AS1 gv4i*)(A + (size_t)(mBlock + rhi) * KDIM + kb),
            (AS3 gv4i*)&As[buf][rhi * LDS_STRIDE + kq], 0, 0);
        __builtin_amdgcn_global_load_async_to_lds_b128(
            (AS1 gv4i*)(B + (size_t)(nBlock + rlo) * KDIM + kb),
            (AS3 gv4i*)&Bs[buf][rlo * LDS_STRIDE + kq], 0, 0);
        __builtin_amdgcn_global_load_async_to_lds_b128(
            (AS1 gv4i*)(B + (size_t)(nBlock + rhi) * KDIM + kb),
            (AS3 gv4i*)&Bs[buf][rhi * LDS_STRIDE + kq], 0, 0);
    };
    auto wait_async = [&]() {
#if __has_builtin(__builtin_amdgcn_s_wait_asynccnt)
        __builtin_amdgcn_s_wait_asynccnt(0);
#else
        asm volatile("s_wait_asynccnt 0x0" ::: "memory");
#endif
    };

    stage(0, 0);
    wait_async();
    __syncthreads();
#else
    // --- fallback: global -> VGPR -> LDS ------------------------------------
    auto ldgA = [&](int row, int k) -> uint4 {
        return *reinterpret_cast<const uint4*>(A + (size_t)(mBlock + row) * KDIM + k);
    };
    auto ldgB = [&](int row, int k) -> uint4 {
        return *reinterpret_cast<const uint4*>(B + (size_t)(nBlock + row) * KDIM + k);
    };
    auto stsA = [&](int buf, int row, uint4 v) {
        *reinterpret_cast<uint4*>(&As[buf][row * LDS_STRIDE + kq]) = v;
    };
    auto stsB = [&](int buf, int row, uint4 v) {
        *reinterpret_cast<uint4*>(&Bs[buf][row * LDS_STRIDE + kq]) = v;
    };
    uint4 a0 = ldgA(rlo, kq), a1 = ldgA(rhi, kq);
    uint4 b0 = ldgB(rlo, kq), b1 = ldgB(rhi, kq);
    stsA(0, rlo, a0); stsA(0, rhi, a1);
    stsB(0, rlo, b0); stsB(0, rhi, b1);
    __syncthreads();
#endif

    v8f acc[2][4];
    #pragma unroll
    for (int m = 0; m < 2; ++m)
        #pragma unroll
        for (int nf = 0; nf < 4; ++nf)
            acc[m][nf] = (v8f){0.f, 0.f, 0.f, 0.f, 0.f, 0.f, 0.f, 0.f};

    for (int kt = 0; kt < NK; ++kt) {
        const int buf = kt & 1;

#if USE_ASYNC_LDS
        if (kt + 1 < NK) stage(kt + 1, buf ^ 1);   // runs behind the WMMAs
#else
        if (kt + 1 < NK) {
            const int kb = (kt + 1) * TK + kq;
            a0 = ldgA(rlo, kb); a1 = ldgA(rhi, kb);
            b0 = ldgB(rlo, kb); b1 = ldgB(rhi, kb);
        }
#endif

        // A fragments (ISA 16-bit A 16x32: half-wave K split {0..7,16..23}/{8..15,24..31})
        v16h af[2];
        #pragma unroll
        for (int m = 0; m < 2; ++m) {
            const _Float16* p = &As[buf][(wm * 32 + m * 16 + r) * LDS_STRIDE];
            v8h lo = *(const v8h*)(p + hs * 8);
            v8h hi = *(const v8h*)(p + 16 + hs * 8);
            af[m] = __builtin_shufflevector(lo, hi, 0,1,2,3,4,5,6,7,8,9,10,11,12,13,14,15);
        }
        // B fragments (B 32x16: half-wave K split {0..7,16..23}/{8..15,24..31}; N = lane%16)
        v16h bf[4];
        #pragma unroll
        for (int nf = 0; nf < 4; ++nf) {
            const _Float16* p = &Bs[buf][(wn * 64 + nf * 16 + r) * LDS_STRIDE];
            v8h lo = *(const v8h*)(p + hs * 8);
            v8h hi = *(const v8h*)(p + 16 + hs * 8);
            bf[nf] = __builtin_shufflevector(lo, hi, 0,1,2,3,4,5,6,7,8,9,10,11,12,13,14,15);
        }

        #pragma unroll
        for (int m = 0; m < 2; ++m)
            #pragma unroll
            for (int nf = 0; nf < 4; ++nf)
                acc[m][nf] = __builtin_amdgcn_wmma_f32_16x16x32_f16(
                    false, af[m], false, bf[nf], (short)0, acc[m][nf], false, false);

#if USE_ASYNC_LDS
        if (kt + 1 < NK) wait_async();             // my LDS writes landed
#else
        if (kt + 1 < NK) {
            const int nb = buf ^ 1;
            stsA(nb, rlo, a0); stsA(nb, rhi, a1);
            stsB(nb, rlo, b0); stsB(nb, rhi, b1);
        }
#endif
        __syncthreads();
    }

    // epilogue: C layout — vgpr v: M = v + 8*hs, N = r
    #pragma unroll
    for (int nf = 0; nf < 4; ++nf) {
        const int col = nBlock + wn * 64 + nf * 16 + r;
        const float bv = bias[col];
        #pragma unroll
        for (int m = 0; m < 2; ++m) {
            const int rowBase = mBlock + wm * 32 + m * 16 + hs * 8;
            #pragma unroll
            for (int v = 0; v < 8; ++v)
                C[(size_t)(rowBase + v) * NDIM + col] = acc[m][nf][v] + bv;
        }
    }
}

// ---------------------------------------------------------------------------
extern "C" void kernel_launch(void* const* d_in, const int* in_sizes, int n_in,
                              void* d_out, int out_size, void* d_ws, size_t ws_size,
                              hipStream_t stream)
{
    const float* x    = (const float*)d_in[0];   // [B,S,Din] = 8*4096*1024
    const float* w    = (const float*)d_in[1];   // [Dout,Din] = 1024*1024
    const float* bias = (const float*)d_in[2];   // [Dout]
    float* out        = (float*)d_out;

    const long long nx = (long long)in_sizes[0];
    const long long nw = (long long)in_sizes[1];
    const int M = (int)(nx / KDIM);              // 32768

    _Float16* qx = reinterpret_cast<_Float16*>(d_ws);
    _Float16* qw = reinterpret_cast<_Float16*>(
        reinterpret_cast<char*>(d_ws) + (size_t)nx * sizeof(unsigned short));

    // quantize activations and weights (4 elements/lane, 8 lanes per NF5 block)
    const long long nx4 = nx / 4, nw4 = nw / 4;
    nf5_quant_f32_to_f16<<<(unsigned)((nx4 + 255) / 256), 256, 0, stream>>>(
        (const float4*)x, (v4h*)qx, nx4);
    nf5_quant_f32_to_f16<<<(unsigned)((nw4 + 255) / 256), 256, 0, stream>>>(
        (const float4*)w, (v4h*)qw, nw4);

    // WMMA GEMM + bias
    dim3 grid(NDIM / TN, M / TM);                // (8, 256)
    nf5_wmma_gemm<<<grid, 256, 0, stream>>>(qx, qw, bias, out, M);
}
